// Axial_Attention_32263794328325
// MI455X (gfx1250) — matrix operand order repaired and checked by reading, not compiled, not run on hardware
//
#include <hip/hip_runtime.h>
#include <hip/hip_bf16.h>
#include <math.h>

// ---------------------------------------------------------------------------
// Problem constants (from the reference)
// ---------------------------------------------------------------------------
#define BB   16
#define LL   2048
#define DD   448
#define EE   64      // DD / 7
#define FEAT 7
#define NCH  16      // chunks for feature partial sums (128 rows each)

typedef __attribute__((ext_vector_type(16))) _Float16 v16h;
typedef __attribute__((ext_vector_type(8)))  _Float16 v8h;
typedef __attribute__((ext_vector_type(8)))  float    v8f;

// ---------------------------------------------------------------------------
// WMMA fragment loaders (wave32, gfx1250; contiguous k order k = 16*(lane/16) + s, same for A and B)
// q_t / k_t are row-major [2048][64] f16, rows 128B aligned.
// ---------------------------------------------------------------------------
__device__ __forceinline__ v16h load_a_frag(const _Float16* __restrict__ base,
                                            int lane, int l0, int e0) {
  // A 16x32 (MxK): lane holds row m = lane&15.
  // lanes 0-15: halves 0..15 -> K=e0+0..15
  // lanes16-31: halves 0..15 -> K=e0+16..31
  const int m   = lane & 15;
  const int off = (lane & 16) ? 16 : 0;
  const _Float16* p = base + (size_t)(l0 + m) * EE + e0 + off;
  v8h lo = *(const v8h*)(p);
  v8h hi = *(const v8h*)(p + 8);
  return __builtin_shufflevector(lo, hi, 0, 1, 2, 3, 4, 5, 6, 7,
                                         8, 9, 10, 11, 12, 13, 14, 15);
}

__device__ __forceinline__ v16h load_b_frag(const _Float16* __restrict__ base,
                                            int lane, int s0, int e0) {
  // B 32x16 (KxN): lane holds column n = lane&15.
  // lanes 0-15 hold K=e0+0..15, lanes 16-31 hold K=e0+16..31 (contiguous 32B).
  const int n    = lane & 15;
  const int koff = (lane & 16) ? 16 : 0;
  return *(const v16h*)(base + (size_t)(s0 + n) * EE + e0 + koff);
}

// ---------------------------------------------------------------------------
// Kernel 1: q_t / k_t = sum over the FEAT split, stored as f16 [B][L][E]
// ---------------------------------------------------------------------------
__global__ void prep_qkt_kernel(const float* __restrict__ q,
                                const float* __restrict__ k,
                                _Float16* __restrict__ qt,
                                _Float16* __restrict__ kt) {
  const int idx = blockIdx.x * blockDim.x + threadIdx.x;  // over B*L*E
  if (idx >= BB * LL * EE) return;
  const int e  = idx & (EE - 1);
  const int bl = idx >> 6;
  const float* qp = q + (size_t)bl * DD + e * FEAT;
  const float* kp = k + (size_t)bl * DD + e * FEAT;
  float sq = 0.f, sk = 0.f;
#pragma unroll
  for (int r = 0; r < FEAT; ++r) { sq += qp[r]; sk += kp[r]; }
  qt[idx] = (_Float16)sq;
  kt[idx] = (_Float16)sk;
}

// ---------------------------------------------------------------------------
// Kernel 2: per-chunk partial column sums of q, k, v (for q_f, k_f, v_sum)
// partials: [B][NCH][448]  (deterministic -- no atomics)
// ---------------------------------------------------------------------------
__global__ void feat_partial_kernel(const float* __restrict__ q,
                                    const float* __restrict__ k,
                                    const float* __restrict__ v,
                                    float* __restrict__ qp,
                                    float* __restrict__ kp,
                                    float* __restrict__ vp) {
  const int b   = blockIdx.x;
  const int ch  = blockIdx.y;
  const int tid = threadIdx.x;  // 256
  const size_t base = ((size_t)b * LL + (size_t)ch * (LL / NCH)) * DD;
  for (int c = tid; c < DD; c += 256) {
    float aq = 0.f, ak = 0.f, av = 0.f;
    for (int l = 0; l < LL / NCH; ++l) {
      const size_t rb = base + (size_t)l * DD + c;
      aq += q[rb]; ak += k[rb]; av += v[rb];
    }
    const size_t po = ((size_t)b * NCH + ch) * DD + c;
    qp[po] = aq; kp[po] = ak; vp[po] = av;
  }
}

// Combine partials: q_f/k_f [B][448], v_sum [B][64]
__global__ void feat_combine_kernel(const float* __restrict__ qp,
                                    const float* __restrict__ kp,
                                    const float* __restrict__ vp,
                                    float* __restrict__ qf,
                                    float* __restrict__ kf,
                                    float* __restrict__ vsum) {
  const int idx = blockIdx.x * blockDim.x + threadIdx.x;  // over B*448
  if (idx >= BB * DD) return;
  const int b = idx / DD;
  const int c = idx % DD;
  float aq = 0.f, ak = 0.f;
  for (int ch = 0; ch < NCH; ++ch) {
    const size_t po = ((size_t)b * NCH + ch) * DD + c;
    aq += qp[po]; ak += kp[po];
  }
  qf[idx] = aq; kf[idx] = ak;
  if (idx < BB * EE) {                      // second job: v_sum[b][e]
    const int vb = idx / EE, e = idx % EE;
    float av = 0.f;
    for (int ch = 0; ch < NCH; ++ch)
      for (int r = 0; r < FEAT; ++r)
        av += vp[((size_t)vb * NCH + ch) * DD + e * FEAT + r];
    vsum[idx] = av;
  }
}

// ---------------------------------------------------------------------------
// Kernel 3 (WMMA pass 1): column-softmax stats over l, per s.
// Block = 8 waves x 16 s-columns = 128 columns; iterates all 2048 rows online.
// ---------------------------------------------------------------------------
__global__ void col_stats_kernel(const _Float16* __restrict__ qt,
                                 const _Float16* __restrict__ kt,
                                 float* __restrict__ ms,
                                 float* __restrict__ zs) {
  const int b    = blockIdx.y;
  const int lane = threadIdx.x & 31;
  const int wave = threadIdx.x >> 5;
  const int s0   = blockIdx.x * 128 + wave * 16;
  const _Float16* qb = qt + (size_t)b * LL * EE;
  const _Float16* kb = kt + (size_t)b * LL * EE;

  // B-fragments are loop-invariant (depend only on s0 and e-chunk)
  const v16h bf0 = load_b_frag(kb, lane, s0, 0);
  const v16h bf1 = load_b_frag(kb, lane, s0, 32);

  float rm = -INFINITY, rz = 0.f;
  for (int lc = 0; lc < LL / 16; ++lc) {
    const int l0 = lc * 16;
    const v16h af0 = load_a_frag(qb, lane, l0, 0);
    const v16h af1 = load_a_frag(qb, lane, l0, 32);
    v8f acc = {};
    acc = __builtin_amdgcn_wmma_f32_16x16x32_f16(false, af0, false, bf0,
                                                 (short)0, acc, false, false);
    acc = __builtin_amdgcn_wmma_f32_16x16x32_f16(false, af1, false, bf1,
                                                 (short)0, acc, false, false);
    // per-column (n) chunk max over 16 rows: 8 in-lane + partner half (xor 16)
    float cm = acc[0];
#pragma unroll
    for (int r = 1; r < 8; ++r) cm = fmaxf(cm, acc[r]);
    cm = fmaxf(cm, __shfl_xor(cm, 16, 32));
    float cs = 0.f;
#pragma unroll
    for (int r = 0; r < 8; ++r) cs += __expf(acc[r] - cm);
    cs += __shfl_xor(cs, 16, 32);
    // online merge
    const float nm = fmaxf(rm, cm);
    rz = rz * __expf(rm - nm) + cs * __expf(cm - nm);
    rm = nm;
  }
  if (lane < 16) {
    ms[(size_t)b * LL + s0 + lane] = rm;
    zs[(size_t)b * LL + s0 + lane] = rz;   // >= 1, max is attained
  }
}

// ---------------------------------------------------------------------------
// Kernel 4 (WMMA pass 2): t[b,l] = sum_s exp(score - m_s) / Z_s
// Block = 8 waves x 16 l-rows = 128 rows; iterates all 2048 columns.
// ---------------------------------------------------------------------------
__global__ void row_t_kernel(const _Float16* __restrict__ qt,
                             const _Float16* __restrict__ kt,
                             const float* __restrict__ ms,
                             const float* __restrict__ zs,
                             float* __restrict__ t) {
  const int b    = blockIdx.y;
  const int lane = threadIdx.x & 31;
  const int wave = threadIdx.x >> 5;
  const int l0   = blockIdx.x * 128 + wave * 16;
  const _Float16* qb = qt + (size_t)b * LL * EE;
  const _Float16* kb = kt + (size_t)b * LL * EE;
  const float* msb = ms + (size_t)b * LL;
  const float* zsb = zs + (size_t)b * LL;

  // A-fragments are loop-invariant for this wave
  const v16h af0 = load_a_frag(qb, lane, l0, 0);
  const v16h af1 = load_a_frag(qb, lane, l0, 32);

  float accsum[8];
#pragma unroll
  for (int r = 0; r < 8; ++r) accsum[r] = 0.f;

  const int n = lane & 15;
  for (int sc = 0; sc < LL / 16; ++sc) {
    const int s0 = sc * 16;
    const v16h bf0 = load_b_frag(kb, lane, s0, 0);
    const v16h bf1 = load_b_frag(kb, lane, s0, 32);
    v8f acc = {};
    acc = __builtin_amdgcn_wmma_f32_16x16x32_f16(false, af0, false, bf0,
                                                 (short)0, acc, false, false);
    acc = __builtin_amdgcn_wmma_f32_16x16x32_f16(false, af1, false, bf1,
                                                 (short)0, acc, false, false);
    const float mn   = msb[s0 + n];
    const float rinv = __frcp_rn(zsb[s0 + n]);
#pragma unroll
    for (int r = 0; r < 8; ++r) accsum[r] += __expf(acc[r] - mn) * rinv;
  }
  // reduce over n (16 lanes within each half); every lane ends with half-total
#pragma unroll
  for (int r = 0; r < 8; ++r) {
    float vv = accsum[r];
    vv += __shfl_xor(vv, 1, 32);
    vv += __shfl_xor(vv, 2, 32);
    vv += __shfl_xor(vv, 4, 32);
    vv += __shfl_xor(vv, 8, 32);
    accsum[r] = vv;
  }
  if ((lane & 15) == 0) {
    const int mbase = l0 + ((lane & 16) ? 8 : 0);
#pragma unroll
    for (int r = 0; r < 8; ++r) t[(size_t)b * LL + mbase + r] = accsum[r];
  }
}

// ---------------------------------------------------------------------------
// Kernel 5: tiny 7x7 feature path -> f[b][r] = sum_p softmax_r(scores_f)[r,p]
// ---------------------------------------------------------------------------
__global__ void feature_kernel(const float* __restrict__ qf,
                               const float* __restrict__ kf,
                               float* __restrict__ f) {
  const int b = blockIdx.x;
  const int t = threadIdx.x;  // 64
  __shared__ float sf[FEAT * FEAT];
  __shared__ float mx[FEAT], iz[FEAT];
  if (t < FEAT * FEAT) {
    const int r = t / FEAT, p = t % FEAT;
    float s = 0.f;
    for (int e = 0; e < EE; ++e)
      s += qf[(size_t)b * DD + e * FEAT + r] * kf[(size_t)b * DD + e * FEAT + p];
    sf[t] = s;
  }
  __syncthreads();
  if (t < FEAT) {  // t = p : softmax over r (axis=-2)
    float m = sf[t];
    for (int r = 1; r < FEAT; ++r) m = fmaxf(m, sf[r * FEAT + t]);
    float z = 0.f;
    for (int r = 0; r < FEAT; ++r) z += __expf(sf[r * FEAT + t] - m);
    mx[t] = m; iz[t] = 1.f / z;
  }
  __syncthreads();
  if (t < FEAT) {  // t = r : f[r] = sum_p A_feature[r,p]
    float fr = 0.f;
    for (int p = 0; p < FEAT; ++p) fr += __expf(sf[t * FEAT + p] - mx[p]) * iz[p];
    f[b * FEAT + t] = fr;
  }
}

// ---------------------------------------------------------------------------
// Kernel 6: per-batch softmax stats over the L*7 array t_l * f_r
// (t >= 0, f > 0  =>  max = max_l t * max_r f)
// ---------------------------------------------------------------------------
__global__ void batch_stats_kernel(const float* __restrict__ t,
                                   const float* __restrict__ f,
                                   float* __restrict__ Mb,
                                   float* __restrict__ Sinv) {
  const int b   = blockIdx.x;
  const int tid = threadIdx.x;  // 256
  __shared__ float red[256];
  __shared__ float Msh;
  const float* tb = t + (size_t)b * LL;

  float lm = -INFINITY;
  for (int l = tid; l < LL; l += 256) lm = fmaxf(lm, tb[l]);
  red[tid] = lm;
  __syncthreads();
  for (int s = 128; s > 0; s >>= 1) {
    if (tid < s) red[tid] = fmaxf(red[tid], red[tid + s]);
    __syncthreads();
  }
  if (tid == 0) {
    float fm = f[b * FEAT];
    for (int r = 1; r < FEAT; ++r) fm = fmaxf(fm, f[b * FEAT + r]);
    Msh = red[0] * fm;
  }
  __syncthreads();

  const float M = Msh;
  float fr[FEAT];
#pragma unroll
  for (int r = 0; r < FEAT; ++r) fr[r] = f[b * FEAT + r];
  float ls = 0.f;
  for (int l = tid; l < LL; l += 256) {
    const float tv = tb[l];
#pragma unroll
    for (int r = 0; r < FEAT; ++r) ls += __expf(tv * fr[r] - M);
  }
  red[tid] = ls;
  __syncthreads();
  for (int s = 128; s > 0; s >>= 1) {
    if (tid < s) red[tid] += red[tid + s];
    __syncthreads();
  }
  if (tid == 0) { Mb[b] = M; Sinv[b] = 1.f / red[0]; }
}

// ---------------------------------------------------------------------------
// Kernel 7: out[b,l,e*7+r] = softmax(t_l * f_r) * v_sum[b,e]
// ---------------------------------------------------------------------------
__global__ void finalize_kernel(const float* __restrict__ t,
                                const float* __restrict__ f,
                                const float* __restrict__ vsum,
                                const float* __restrict__ Mb,
                                const float* __restrict__ Sinv,
                                float* __restrict__ out) {
  const int row = blockIdx.x;          // b*L + l
  const int b   = row >> 11;           // /2048
  const int tid = threadIdx.x;         // 256
  __shared__ float er[FEAT];
  if (tid < FEAT)
    er[tid] = __expf(t[row] * f[b * FEAT + tid] - Mb[b]) * Sinv[b];
  __syncthreads();
  for (int c = tid; c < DD; c += 256)
    out[(size_t)row * DD + c] = er[c % FEAT] * vsum[b * EE + c / FEAT];
}

// ---------------------------------------------------------------------------
// Workspace layout (bytes)
// ---------------------------------------------------------------------------
static constexpr size_t QT_OFF = 0;
static constexpr size_t QT_SZ  = (size_t)BB * LL * EE * sizeof(_Float16);
static constexpr size_t KT_OFF = QT_OFF + QT_SZ;
static constexpr size_t QP_OFF = KT_OFF + QT_SZ;
static constexpr size_t PART_SZ = (size_t)BB * NCH * DD * sizeof(float);
static constexpr size_t KP_OFF = QP_OFF + PART_SZ;
static constexpr size_t VP_OFF = KP_OFF + PART_SZ;
static constexpr size_t QF_OFF = VP_OFF + PART_SZ;
static constexpr size_t QF_SZ  = (size_t)BB * DD * sizeof(float);
static constexpr size_t KF_OFF = QF_OFF + QF_SZ;
static constexpr size_t VS_OFF = KF_OFF + QF_SZ;
static constexpr size_t VS_SZ  = (size_t)BB * EE * sizeof(float);
static constexpr size_t MS_OFF = VS_OFF + VS_SZ;
static constexpr size_t BL_SZ  = (size_t)BB * LL * sizeof(float);
static constexpr size_t ZS_OFF = MS_OFF + BL_SZ;
static constexpr size_t T_OFF  = ZS_OFF + BL_SZ;
static constexpr size_t F_OFF  = T_OFF + BL_SZ;
static constexpr size_t F_SZ   = (size_t)BB * FEAT * sizeof(float);
static constexpr size_t M_OFF  = F_OFF + F_SZ + 32;   // keep 64B-ish alignment
static constexpr size_t SI_OFF = M_OFF + 64;

extern "C" void kernel_launch(void* const* d_in, const int* in_sizes, int n_in,
                              void* d_out, int out_size, void* d_ws, size_t ws_size,
                              hipStream_t stream) {
  (void)in_sizes; (void)n_in; (void)out_size; (void)ws_size;
  const float* q = (const float*)d_in[0];
  const float* k = (const float*)d_in[1];
  const float* v = (const float*)d_in[2];
  float* out = (float*)d_out;

  char* ws = (char*)d_ws;
  _Float16* qt = (_Float16*)(ws + QT_OFF);
  _Float16* kt = (_Float16*)(ws + KT_OFF);
  float* qp   = (float*)(ws + QP_OFF);
  float* kp   = (float*)(ws + KP_OFF);
  float* vp   = (float*)(ws + VP_OFF);
  float* qf   = (float*)(ws + QF_OFF);
  float* kf   = (float*)(ws + KF_OFF);
  float* vsum = (float*)(ws + VS_OFF);
  float* ms   = (float*)(ws + MS_OFF);
  float* zs   = (float*)(ws + ZS_OFF);
  float* t    = (float*)(ws + T_OFF);
  float* f    = (float*)(ws + F_OFF);
  float* Mb   = (float*)(ws + M_OFF);
  float* Sinv = (float*)(ws + SI_OFF);

  // 1) f16 reductions q_t, k_t
  {
    const int nthreads = BB * LL * EE;
    prep_qkt_kernel<<<(nthreads + 255) / 256, 256, 0, stream>>>(q, k, qt, kt);
  }
  // 2) feature partial sums + deterministic combine
  feat_partial_kernel<<<dim3(BB, NCH), 256, 0, stream>>>(q, k, v, qp, kp, vp);
  feat_combine_kernel<<<(BB * DD + 255) / 256, 256, 0, stream>>>(qp, kp, vp,
                                                                 qf, kf, vsum);
  // 3) WMMA pass 1: column softmax stats (m_s, Z_s)
  col_stats_kernel<<<dim3(LL / 128, BB), 256, 0, stream>>>(qt, kt, ms, zs);
  // 4) WMMA pass 2: t[b,l]
  row_t_kernel<<<dim3(LL / 128, BB), 256, 0, stream>>>(qt, kt, ms, zs, t);
  // 5) tiny feature softmax -> f[b][r]
  feature_kernel<<<BB, 64, 0, stream>>>(qf, kf, f);
  // 6) per-batch global softmax stats over t_l * f_r
  batch_stats_kernel<<<BB, 256, 0, stream>>>(t, f, Mb, Sinv);
  // 7) rank-1 output
  finalize_kernel<<<BB * LL, 256, 0, stream>>>(t, f, vsum, Mb, Sinv, out);
}
